// GATv2_42150809043286
// MI455X (gfx1250) — matrix operand order repaired and checked by reading, not compiled, not run on hardware
//
#include <hip/hip_runtime.h>
#include <hip/hip_bf16.h>

// ---------------------------------------------------------------------------
// GATv2 (2 layers) on gfx1250.
// GEMMs -> v_wmma_f32_16x16x32_bf16 (f32 accumulate), fully unrolled K=256,
//          pre-transposed bf16 weights so all fragment loads are b128 vectors.
// Edge softmax / aggregation -> L2-resident gathers + float atomics.
// ---------------------------------------------------------------------------

#define N_NODES   50000
#define N_EDGES   800000
#define F_IN      256
#define F_HID     256      /* 4 heads * 64 */
#define H1        4
#define D1        64
#define D2        40
#define D2_PAD    48       /* pad layer-2 columns to 3 full 16-wide tiles */
#define NEG_SLOPE 0.2f

typedef __bf16 bf16_t;
typedef __attribute__((ext_vector_type(16))) __bf16 v16bf;
typedef __attribute__((ext_vector_type(8)))  __bf16 v8bf;
typedef __attribute__((ext_vector_type(8)))  float  v8f;

// order-preserving float->uint encoding for atomic segment-max
__device__ __forceinline__ unsigned fenc(float f) {
  unsigned u = __float_as_uint(f);
  return (u & 0x80000000u) ? ~u : (u | 0x80000000u);
}
__device__ __forceinline__ float fdec(unsigned u) {
  u = (u & 0x80000000u) ? (u & 0x7FFFFFFFu) : ~u;
  return __uint_as_float(u);
}

// ---------------------------------------------------------------------------
__global__ void k_zero(float* __restrict__ p, int n) {
  int i = blockIdx.x * blockDim.x + threadIdx.x;
  if (i < n) p[i] = 0.0f;
}

__global__ void k_cvt_bf16(const float* __restrict__ in, bf16_t* __restrict__ out, int n) {
  int i = blockIdx.x * blockDim.x + threadIdx.x;
  if (i < n) out[i] = (bf16_t)in[i];
}

// out[n*rows + k] = (n < cols_in) ? bf16(in[k*cols_in + n]) : 0   (transpose+pad)
__global__ void k_cvt_transpose_bf16(const float* __restrict__ in,
                                     bf16_t* __restrict__ out,
                                     int rows, int cols_in, int cols_pad) {
  int i = blockIdx.x * blockDim.x + threadIdx.x;
  if (i >= cols_pad * rows) return;
  int n = i / rows, k = i - n * rows;
  float v = (n < cols_in) ? in[(size_t)k * cols_in + n] : 0.0f;
  out[i] = (bf16_t)v;
}

// ---------------------------------------------------------------------------
// C[M, ncols_out] = A[M, K] (bf16 row-major) * Bt (bf16 COLUMN-major: [n][K])
// One wave per 16x16 output tile; K templated so the loop fully unrolls into
// 8 v_wmma_f32_16x16x32_bf16 with unconditional b128 fragment loads.
// Fragment layouts per CDNA5 ISA 7.12.2:
//   A 16x32 bf16: lane L holds row M=L%16; lanes 0-15 K=0..7 & 16..23,
//                 lanes 16-31 K=8..15 & 24..31.
//   B 32x16 bf16: lane L holds col N=L%16; this code (A too) uses lanes 0-15 K=0..15,
//                 lanes 16-31 K=16..31  -> 16 contiguous bf16 from Bt column.
//   C 16x16 f32:  lane L holds col N=L%16; VGPR r -> row r (+8 upper half).
template <int K>
__global__ void k_gemm_bf16(const bf16_t* __restrict__ A,
                            const bf16_t* __restrict__ Bt,
                            float* __restrict__ C,
                            int ncols_out, int n_tiles) {
  const int wave  = threadIdx.x >> 5;                         // wave-uniform
  const int ntile = blockIdx.y * (blockDim.x >> 5) + wave;
  if (ntile >= n_tiles) return;                               // wave-uniform guard

  const int lane = threadIdx.x & 31;
  const int half = lane >> 4;                                 // 0 or 1
  const int mrow = blockIdx.x * 16 + (lane & 15);
  const int ncol = ntile * 16 + (lane & 15);

  const bf16_t* arow = A  + (size_t)mrow * K;
  const bf16_t* bcol = Bt + (size_t)ncol * K;

  v8f c = {};
#pragma unroll
  for (int k = 0; k < K; k += 32) {
    // A fragment: two 16B contiguous loads, 16 contiguous K per half
    const bf16_t* ap = arow + k + half * 16;
    v8bf alo = *(const v8bf*)ap;
    v8bf ahi = *(const v8bf*)(ap + 8);
    v16bf a = __builtin_shufflevector(alo, ahi,
                0,1,2,3,4,5,6,7,8,9,10,11,12,13,14,15);
    // B fragment: 16 contiguous bf16 from the transposed column (32B load)
    v16bf b = *(const v16bf*)(bcol + k + half * 16);

    c = __builtin_amdgcn_wmma_f32_16x16x32_bf16(
            /*neg_a=*/false, a, /*neg_b=*/false, b,
            /*c_mod=*/(short)0, c, /*reuse_a=*/false, /*reuse_b=*/false);
  }

  const int rbase = blockIdx.x * 16 + half * 8;
  const bool nvalid = (ncol < ncols_out);
#pragma unroll
  for (int r = 0; r < 8; ++r)
    if (nvalid) C[(size_t)(rbase + r) * ncols_out + ncol] = c[r];
}

// ---------------------------------------------------------------------------
// Edge phase (generic over H heads of width D, feature stride = H*D)
// score[e,h] = attn[h,:] . leaky_relu(feat[src]+feat[dst]) ; atomic max -> mx
__global__ void k_score(const float* __restrict__ feat,
                        const int* __restrict__ src, const int* __restrict__ dst,
                        const float* __restrict__ attn,
                        float* __restrict__ score, unsigned* __restrict__ mx,
                        int H, int D, int total) {
  int idx = blockIdx.x * blockDim.x + threadIdx.x;
  if (idx >= total) return;
  int e = idx / H, h = idx - e * H;
  int s = src[e], d = dst[e];
  const float* fs = feat + (size_t)s * (H * D) + h * D;
  const float* fd = feat + (size_t)d * (H * D) + h * D;
  const float* at = attn + h * D;
  float acc = 0.0f;
  for (int i = 0; i < D; ++i) {
    float v = fs[i] + fd[i];
    v = (v > 0.0f) ? v : NEG_SLOPE * v;
    acc += at[i] * v;
  }
  score[idx] = acc;
  atomicMax(mx + d * H + h, fenc(acc));
}

// ex = exp(score - m[dst]); denom[dst] += ex
__global__ void k_expsum(const int* __restrict__ dst,
                         float* __restrict__ score,
                         const unsigned* __restrict__ mx,
                         float* __restrict__ den, int H, int total) {
  int idx = blockIdx.x * blockDim.x + threadIdx.x;
  if (idx >= total) return;
  int e = idx / H, h = idx - e * H;
  int d = dst[e];
  float ex = expf(score[idx] - fdec(mx[d * H + h]));
  score[idx] = ex;
  atomicAdd(den + d * H + h, ex);
}

// alpha = ex / denom[dst]
__global__ void k_alpha(const int* __restrict__ dst, float* __restrict__ score,
                        const float* __restrict__ den, int H, int total) {
  int idx = blockIdx.x * blockDim.x + threadIdx.x;
  if (idx >= total) return;
  int e = idx / H, h = idx - e * H;
  score[idx] = score[idx] / den[dst[e] * H + h];
}

// out[dst, c] += alpha[e, c/D] * feat[src, c]   (c in [0, H*D))
__global__ void k_aggregate(const float* __restrict__ feat,
                            const int* __restrict__ src, const int* __restrict__ dst,
                            const float* __restrict__ alpha,
                            float* __restrict__ out,
                            int H, int D, long total) {
  long idx = (long)blockIdx.x * blockDim.x + threadIdx.x;
  if (idx >= total) return;
  int F = H * D;
  int e = (int)(idx / F);
  int c = (int)(idx - (long)e * F);
  int h = c / D;
  float a = alpha[e * H + h];
  atomicAdd(out + (size_t)dst[e] * F + c, a * feat[(size_t)src[e] * F + c]);
}

// elu + bf16 convert (layer-1 output -> layer-2 GEMM input)
__global__ void k_elu_cvt(const float* __restrict__ in, bf16_t* __restrict__ out, int n) {
  int i = blockIdx.x * blockDim.x + threadIdx.x;
  if (i < n) {
    float v = in[i];
    v = (v > 0.0f) ? v : (expf(v) - 1.0f);
    out[i] = (bf16_t)v;
  }
}

// ---------------------------------------------------------------------------
// Workspace layout (bytes, all 64B-aligned)
static const size_t OFF_XB     = 0;           // bf16 x / later h1b (25.6 MB)
static const size_t OFF_W1T    = 25600000;    // bf16 W1^T [256][256] (128 KB)
static const size_t OFF_W2T    = 25731072;    // bf16 W2^T [48][256]  (24 KB)
static const size_t OFF_FEAT1  = 25755648;    // f32 feat1 (51.2 MB) / later feat2 (8 MB)
static const size_t OFF_OUT1   = 76955648;    // f32 out1 (51.2 MB)
static const size_t OFF_SCORE1 = 128155648;   // f32 [E,4] (12.8 MB)
static const size_t OFF_M1     = 140955648;   // u32 [N,4]
static const size_t OFF_DEN1   = 141755648;   // f32 [N,4]
static const size_t OFF_M2     = 142555648;   // u32 [N]
static const size_t OFF_DEN2   = 142755648;   // f32 [N]
static const size_t OFF_SCORE2 = 142955648;   // f32 [E]

static inline unsigned nblk(long n) { return (unsigned)((n + 255) / 256); }

extern "C" void kernel_launch(void* const* d_in, const int* in_sizes, int n_in,
                              void* d_out, int out_size, void* d_ws, size_t ws_size,
                              hipStream_t stream) {
  const float* x     = (const float*)d_in[0];
  const int*   src   = (const int*)d_in[1];
  const int*   dst   = (const int*)d_in[2];
  const float* W1    = (const float*)d_in[3];
  const float* attn1 = (const float*)d_in[4];
  const float* W2    = (const float*)d_in[5];
  const float* attn2 = (const float*)d_in[6];
  float* out = (float*)d_out;
  char* ws = (char*)d_ws;

  bf16_t*   xb    = (bf16_t*)(ws + OFF_XB);
  bf16_t*   h1b   = (bf16_t*)(ws + OFF_XB);      // reuse after GEMM1 consumed xb
  bf16_t*   w1t   = (bf16_t*)(ws + OFF_W1T);
  bf16_t*   w2t   = (bf16_t*)(ws + OFF_W2T);
  float*    feat1 = (float*)(ws + OFF_FEAT1);
  float*    feat2 = (float*)(ws + OFF_FEAT1);    // reuse after layer 1 done
  float*    out1  = (float*)(ws + OFF_OUT1);
  float*    sc1   = (float*)(ws + OFF_SCORE1);
  unsigned* m1    = (unsigned*)(ws + OFF_M1);
  float*    den1  = (float*)(ws + OFF_DEN1);
  unsigned* m2    = (unsigned*)(ws + OFF_M2);
  float*    den2  = (float*)(ws + OFF_DEN2);
  float*    sc2   = (float*)(ws + OFF_SCORE2);

  // ---- init accumulators (0 bits < fenc(-inf): identity for atomic max)
  k_zero<<<nblk(N_NODES * F_HID), 256, 0, stream>>>(out1, N_NODES * F_HID);
  k_zero<<<nblk(N_NODES * H1),    256, 0, stream>>>((float*)m1,  N_NODES * H1);
  k_zero<<<nblk(N_NODES * H1),    256, 0, stream>>>(den1,        N_NODES * H1);
  k_zero<<<nblk(N_NODES),         256, 0, stream>>>((float*)m2,  N_NODES);
  k_zero<<<nblk(N_NODES),         256, 0, stream>>>(den2,        N_NODES);
  k_zero<<<nblk(N_NODES * D2),    256, 0, stream>>>(out,         N_NODES * D2);

  // ---- bf16 conversions (weights transposed+padded to column-major)
  k_cvt_bf16<<<nblk(N_NODES * F_IN), 256, 0, stream>>>(x, xb, N_NODES * F_IN);
  k_cvt_transpose_bf16<<<nblk(F_HID * F_IN), 256, 0, stream>>>(W1, w1t,
                                                               F_IN, F_HID, F_HID);
  k_cvt_transpose_bf16<<<nblk(D2_PAD * F_HID), 256, 0, stream>>>(W2, w2t,
                                                                 F_HID, D2, D2_PAD);

  // ---- layer 1: feat1 = x @ W1   (3125 x 16 tiles, 4 waves/block)
  k_gemm_bf16<F_IN><<<dim3(N_NODES / 16, 4), 128, 0, stream>>>(xb, w1t, feat1,
                                                               F_HID, F_HID / 16);
  // edge softmax + aggregation
  k_score<<<nblk((long)N_EDGES * H1), 256, 0, stream>>>(feat1, src, dst, attn1,
                                                        sc1, m1, H1, D1, N_EDGES * H1);
  k_expsum<<<nblk((long)N_EDGES * H1), 256, 0, stream>>>(dst, sc1, m1, den1,
                                                         H1, N_EDGES * H1);
  k_alpha<<<nblk((long)N_EDGES * H1), 256, 0, stream>>>(dst, sc1, den1,
                                                        H1, N_EDGES * H1);
  k_aggregate<<<nblk((long)N_EDGES * F_HID), 256, 0, stream>>>(feat1, src, dst, sc1,
                                                               out1, H1, D1,
                                                               (long)N_EDGES * F_HID);
  // elu + convert for layer 2
  k_elu_cvt<<<nblk(N_NODES * F_HID), 256, 0, stream>>>(out1, h1b, N_NODES * F_HID);

  // ---- layer 2: feat2 = h1 @ W2  (3 column tiles, cols 40..47 padded in Bt,
  //      stores guarded to 40)
  k_gemm_bf16<F_HID><<<dim3(N_NODES / 16, 1), 128, 0, stream>>>(h1b, w2t, feat2,
                                                                D2, 3);
  k_score<<<nblk(N_EDGES), 256, 0, stream>>>(feat2, src, dst, attn2,
                                             sc2, m2, 1, D2, N_EDGES);
  k_expsum<<<nblk(N_EDGES), 256, 0, stream>>>(dst, sc2, m2, den2, 1, N_EDGES);
  k_alpha<<<nblk(N_EDGES), 256, 0, stream>>>(dst, sc2, den2, 1, N_EDGES);
  // H=1 so mean over heads == identity; accumulate straight into d_out
  k_aggregate<<<nblk((long)N_EDGES * D2), 256, 0, stream>>>(feat2, src, dst, sc2,
                                                            out, 1, D2,
                                                            (long)N_EDGES * D2);
}
